// Lora_Quant_Linear_13503377178797
// MI455X (gfx1250) — matrix operand order repaired and checked by reading, not compiled, not run on hardware
//
#include <hip/hip_runtime.h>
#include <hip/hip_bf16.h>
#include <stdint.h>

typedef __attribute__((ext_vector_type(16))) __bf16 v16bf;
typedef __attribute__((ext_vector_type(8)))  float  v8f;

// NF4 codebook
__device__ const float NF4_TAB[16] = {
    -1.0f, -0.6961928009986877f, -0.5250730514526367f, -0.39491748809814453f,
    -0.28444138169288635f, -0.18477343022823334f, -0.09105003625154495f, 0.0f,
    0.07958029955625534f, 0.16093020141124725f, 0.24611230194568634f, 0.33791524171829224f,
    0.44070982933044434f, 0.5626170039176941f, 0.7229568362236023f, 1.0f};

#define TOKENS 4096
#define INDIM  4096
#define OUTDIM 4096
#define RANK   16
#define LORA_SCALE 2.0f

#if __has_builtin(__builtin_amdgcn_global_load_async_to_lds_b128) && \
    __has_builtin(__builtin_amdgcn_s_wait_asynccnt)
#define USE_ASYNC_LDS 1
#else
#define USE_ASYNC_LDS 0
#endif

__device__ __forceinline__ unsigned short f2bf(float f) {
    unsigned u = __float_as_uint(f);
    u += 0x7fffu + ((u >> 16) & 1u);   // round-to-nearest-even
    return (unsigned short)(u >> 16);
}

__device__ __forceinline__ uint4 pack8bf(const float* w) {
    uint4 r;
    r.x = (unsigned)f2bf(w[0]) | ((unsigned)f2bf(w[1]) << 16);
    r.y = (unsigned)f2bf(w[2]) | ((unsigned)f2bf(w[3]) << 16);
    r.z = (unsigned)f2bf(w[4]) | ((unsigned)f2bf(w[5]) << 16);
    r.w = (unsigned)f2bf(w[6]) | ((unsigned)f2bf(w[7]) << 16);
    return r;
}

// ---------------- Kernel 1: x fp32 -> bf16 ----------------
__global__ __launch_bounds__(256) void convert_x_bf16(const float* __restrict__ x,
                                                      unsigned short* __restrict__ xb) {
    size_t t = (size_t)blockIdx.x * blockDim.x + threadIdx.x;
    size_t base = t * 8;
    float4 a0 = *reinterpret_cast<const float4*>(x + base);
    float4 a1 = *reinterpret_cast<const float4*>(x + base + 4);
    float w[8] = {a0.x, a0.y, a0.z, a0.w, a1.x, a1.y, a1.z, a1.w};
    *reinterpret_cast<uint4*>(xb + base) = pack8bf(w);
}

// ------- Kernel 2: W' = NF4[codes]*scale + 2*(B@A), bf16 out -------
__global__ __launch_bounds__(256) void dequant_lora_w(const int* __restrict__ codes,
                                                      const float* __restrict__ scales,
                                                      const float* __restrict__ lora_A,
                                                      const float* __restrict__ lora_B,
                                                      unsigned short* __restrict__ wb) {
    size_t t = (size_t)blockIdx.x * blockDim.x + threadIdx.x;
    size_t base = t * 8;                    // 8 consecutive elems, same row, same 64-block
    int o = (int)(base / INDIM);
    int i = (int)(base % INDIM);
    float s = scales[base >> 6];

    int4 c0 = *reinterpret_cast<const int4*>(codes + base);
    int4 c1 = *reinterpret_cast<const int4*>(codes + base + 4);
    float w[8];
    w[0] = NF4_TAB[c0.x & 15] * s;  w[1] = NF4_TAB[c0.y & 15] * s;
    w[2] = NF4_TAB[c0.z & 15] * s;  w[3] = NF4_TAB[c0.w & 15] * s;
    w[4] = NF4_TAB[c1.x & 15] * s;  w[5] = NF4_TAB[c1.y & 15] * s;
    w[6] = NF4_TAB[c1.z & 15] * s;  w[7] = NF4_TAB[c1.w & 15] * s;

    // LoRA fold: w[j] += 2 * sum_r B[o,r] * A[r, i+j]
    float br[RANK];
#pragma unroll
    for (int r4 = 0; r4 < RANK / 4; ++r4) {
        float4 b4 = *reinterpret_cast<const float4*>(lora_B + (size_t)o * RANK + r4 * 4);
        br[r4 * 4 + 0] = b4.x; br[r4 * 4 + 1] = b4.y;
        br[r4 * 4 + 2] = b4.z; br[r4 * 4 + 3] = b4.w;
    }
#pragma unroll
    for (int r = 0; r < RANK; ++r) {
        const float* ap = lora_A + (size_t)r * INDIM + i;
        float4 a0 = *reinterpret_cast<const float4*>(ap);
        float4 a1 = *reinterpret_cast<const float4*>(ap + 4);
        float bscale = LORA_SCALE * br[r];
        w[0] = fmaf(bscale, a0.x, w[0]); w[1] = fmaf(bscale, a0.y, w[1]);
        w[2] = fmaf(bscale, a0.z, w[2]); w[3] = fmaf(bscale, a0.w, w[3]);
        w[4] = fmaf(bscale, a1.x, w[4]); w[5] = fmaf(bscale, a1.y, w[5]);
        w[6] = fmaf(bscale, a1.z, w[6]); w[7] = fmaf(bscale, a1.w, w[7]);
    }
    *reinterpret_cast<uint4*>(wb + base) = pack8bf(w);
}

// ---------------- Kernel 3: out = Xb @ Wb^T via WMMA bf16 ----------------
// Block tile 128(M) x 128(N), K step 32, double-buffered LDS filled by
// global_load_async_to_lds_b128 (ASYNCcnt). 8 waves: 2 along M x 4 along N,
// each wave owns 4x2 tiles of 16x16 f32 accumulators.
#define M_TILE 128
#define N_TILE 128
#define K_TILE 32
#define LDSPAD 40   // bf16 row pitch: 80B = 20 banks -> conflict-free, 16B aligned

union FragBF {
    uint4 u4[2];
    v16bf v;
};

#if USE_ASYNC_LDS
typedef int v4i_ __attribute__((vector_size(16)));
__device__ __forceinline__ void async_b128(const void* g, void* l) {
    __builtin_amdgcn_global_load_async_to_lds_b128(
        (__attribute__((address_space(1))) v4i_*)(uintptr_t)g,
        (__attribute__((address_space(3))) v4i_*)(uintptr_t)l,
        0, 0);
}
#endif

__global__ __launch_bounds__(256, 2) void gemm_bf16_wmma(const unsigned short* __restrict__ Xb,
                                                         const unsigned short* __restrict__ Wb,
                                                         float* __restrict__ out) {
    __shared__ unsigned short xs[2][M_TILE][LDSPAD];
    __shared__ unsigned short wsh[2][N_TILE][LDSPAD];

    const int tid  = threadIdx.x;
    const int lane = tid & 31;
    const int wave = tid >> 5;
    const int wave_m = wave & 1;   // 0..1  -> 64-row slab
    const int wave_n = wave >> 1;  // 0..3  -> 32-col slab

    const int m_blk = blockIdx.y * M_TILE;
    const int n_blk = blockIdx.x * N_TILE;

    // loader mapping: 256 threads, each covers 16 bf16 (32B) of one tile row per K step
    const int lr = tid >> 1;          // 0..127
    const int lc = (tid & 1) * 16;    // 0 or 16

    const unsigned short* xrow = Xb + (size_t)(m_blk + lr) * INDIM + lc;
    const unsigned short* wrow = Wb + (size_t)(n_blk + lr) * INDIM + lc;

    v8f acc[4][2];
#pragma unroll
    for (int i = 0; i < 4; ++i)
#pragma unroll
        for (int j = 0; j < 2; ++j)
            acc[i][j] = (v8f)(0.0f);

    // Fragment addressing per ISA VGPR layouts (wave32):
    // A (16x32 bf16): lanes 0-15 row M=lane, K chunks {0..7,16..23}; lanes 16-31 chunks {8..15,24..31}
    const int a_row  = wave_m * 64 + (lane & 15);
    const int a_koff = (lane < 16) ? 0 : 8;
    // B (32x16 bf16): col N = lane&15; lanes 0-15 K {0..7,16..23}, lanes 16-31 K {8..15,24..31}
    const int b_koff = (lane < 16) ? 0 : 8;
    const int b_coll = lane & 15;

#if USE_ASYNC_LDS
    // Prologue: async-fill buffer 0
    async_b128(xrow,     &xs[0][lr][lc]);
    async_b128(xrow + 8, &xs[0][lr][lc + 8]);
    async_b128(wrow,     &wsh[0][lr][lc]);
    async_b128(wrow + 8, &wsh[0][lr][lc + 8]);
#endif

    for (int k0 = 0; k0 < INDIM; k0 += K_TILE) {
#if USE_ASYNC_LDS
        const int cur = (k0 / K_TILE) & 1;
        __builtin_amdgcn_s_wait_asynccnt(0);  // this wave's fills of buf[cur] done
        __syncthreads();                      // all waves' fills done; prior reads of buf[cur^1] done
        if (k0 + K_TILE < INDIM) {
            const int nxt = cur ^ 1;
            const unsigned short* xsk = xrow + k0 + K_TILE;
            const unsigned short* wsk = wrow + k0 + K_TILE;
            async_b128(xsk,     &xs[nxt][lr][lc]);
            async_b128(xsk + 8, &xs[nxt][lr][lc + 8]);
            async_b128(wsk,     &wsh[nxt][lr][lc]);
            async_b128(wsk + 8, &wsh[nxt][lr][lc + 8]);
            // keep the L2 stream warm a few tiles ahead
            __builtin_prefetch(xsk + 8 * K_TILE, 0, 0);
            __builtin_prefetch(wsk + 8 * K_TILE, 0, 0);
        }
#else
        const int cur = 0;
        uint4 xv0 = *reinterpret_cast<const uint4*>(xrow + k0);
        uint4 xv1 = *reinterpret_cast<const uint4*>(xrow + k0 + 8);
        uint4 wv0 = *reinterpret_cast<const uint4*>(wrow + k0);
        uint4 wv1 = *reinterpret_cast<const uint4*>(wrow + k0 + 8);
        __syncthreads();
        *reinterpret_cast<uint4*>(&xs[0][lr][lc])      = xv0;
        *reinterpret_cast<uint4*>(&xs[0][lr][lc + 8])  = xv1;
        *reinterpret_cast<uint4*>(&wsh[0][lr][lc])     = wv0;
        *reinterpret_cast<uint4*>(&wsh[0][lr][lc + 8]) = wv1;
        __syncthreads();
#endif

        FragBF a[4];
#pragma unroll
        for (int i = 0; i < 4; ++i) {
            const unsigned short* p = &xs[cur][a_row + i * 16][a_koff];
            a[i].u4[0] = *reinterpret_cast<const uint4*>(p);        // K = koff..koff+7
            a[i].u4[1] = *reinterpret_cast<const uint4*>(p + 16);   // K = koff+16..koff+23
        }
        FragBF b[2];
#pragma unroll
        for (int j = 0; j < 2; ++j) {
            const unsigned short* p = &wsh[cur][wave_n * 32 + j * 16 + b_coll][b_koff];
            b[j].u4[0] = *reinterpret_cast<const uint4*>(p);
            b[j].u4[1] = *reinterpret_cast<const uint4*>(p + 16);
        }
#pragma unroll
        for (int i = 0; i < 4; ++i)
#pragma unroll
            for (int j = 0; j < 2; ++j)
                acc[i][j] = __builtin_amdgcn_wmma_f32_16x16x32_bf16(
                    false, a[i].v, false, b[j].v, (short)0, acc[i][j], false, false);
    }

    // Epilogue: C/D layout — VGPR r: lanes 0-15 (M=r, N=lane), lanes 16-31 (M=r+8, N=lane-16)
    const int n_out = n_blk + wave_n * 32 + (lane & 15);
    const int m_hi  = (lane < 16) ? 0 : 8;
#pragma unroll
    for (int i = 0; i < 4; ++i) {
#pragma unroll
        for (int j = 0; j < 2; ++j) {
            const int n = n_out + j * 16;
            const int m0 = m_blk + wave_m * 64 + i * 16 + m_hi;
#pragma unroll
            for (int r = 0; r < 8; ++r) {
                out[(size_t)(m0 + r) * OUTDIM + n] = acc[i][j][r];
            }
        }
    }
}

extern "C" void kernel_launch(void* const* d_in, const int* in_sizes, int n_in,
                              void* d_out, int out_size, void* d_ws, size_t ws_size,
                              hipStream_t stream) {
    const float* x      = (const float*)d_in[0];
    const int*   codes  = (const int*)d_in[1];
    const float* scales = (const float*)d_in[2];
    const float* lora_A = (const float*)d_in[3];
    const float* lora_B = (const float*)d_in[4];
    float* out = (float*)d_out;

    unsigned short* xb = (unsigned short*)d_ws;                                       // 32 MB
    unsigned short* wb = (unsigned short*)((char*)d_ws + (size_t)TOKENS * INDIM * 2); // 32 MB

    {
        int total = TOKENS * INDIM;
        int blocks = total / (256 * 8);
        convert_x_bf16<<<blocks, 256, 0, stream>>>(x, xb);
    }
    {
        int total = OUTDIM * INDIM;
        int blocks = total / (256 * 8);
        dequant_lora_w<<<blocks, 256, 0, stream>>>(codes, scales, lora_A, lora_B, wb);
    }
    {
        dim3 grid(OUTDIM / N_TILE, TOKENS / M_TILE);
        gemm_bf16_wmma<<<grid, 256, 0, stream>>>(xb, wb, out);
    }
}
